// SpikMambaBlock_16836271800584
// MI455X (gfx1250) — matrix operand order repaired and checked by reading, not compiled, not run on hardware
//
#include <hip/hip_runtime.h>
#include <hip/hip_bf16.h>

typedef _Float16 half8  __attribute__((ext_vector_type(8)));
typedef _Float16 half16 __attribute__((ext_vector_type(16)));
typedef float    float8 __attribute__((ext_vector_type(8)));

#define T_  4
#define B_  4
#define C_  128
#define H_  48
#define W_  48
#define HW_ (H_*W_)      // 2304
#define TB_ (T_*B_)      // 16  (== G groups in ss2d)
#define P_  (TB_*HW_)    // 36864 pixel rows
#define K_  4            // scan directions
#define R_  8
#define EC_ (4*C_)       // 512
#define PH_ (H_+2)       // 50 (zero-padded halo for conv)
#define PW_ (W_+2)       // 50

enum { EPI_F32_PC = 0, EPI_F16_PC = 1, EPI_F16_PC_SILU = 2, EPI_F32_CHW_ADD = 3 };

__device__ __forceinline__ float silu_f(float x)     { return x / (1.0f + __expf(-x)); }
__device__ __forceinline__ float softplus_f(float x) { return (x > 20.0f) ? x : log1pf(__expf(x)); }

// s(k,l): position in the HW-ordered image that direction-k reads at scan step l.
// All four maps are involutions, so scan outputs scatter back with the same map.
__device__ __forceinline__ int map_s(int k, int l) {
  int m = (k >= 2) ? (HW_ - 1 - l) : l;
  return (k & 1) ? ((m % H_) * W_ + (m / H_)) : m;
}

__device__ __forceinline__ float8 wmma16(half16 a, half16 b, float8 c) {
  return __builtin_amdgcn_wmma_f32_16x16x32_f16(false, a, false, b, (short)0, c, false, false);
}
__device__ __forceinline__ half16 mk_a(const _Float16* p) {
  half8 lo = *(const half8*)(p);
  half8 hi = *(const half8*)(p + 8);
  return __builtin_shufflevector(lo, hi, 0,1,2,3,4,5,6,7,8,9,10,11,12,13,14,15);
}

// ---------------- fills / packing ----------------
__global__ void k_zero_h8(_Float16* __restrict__ p, int n8) {   // n8 = count of half8 chunks
  int i = blockIdx.x * blockDim.x + threadIdx.x;
  if (i < n8) { half8 z = {}; *(half8*)(p + (size_t)i * 8) = z; }
}

__global__ void k_pack_f16(const float* __restrict__ in, _Float16* __restrict__ out, int n) {
  int i = blockIdx.x * blockDim.x + threadIdx.x;
  if (i < n) out[i] = (_Float16)in[i];
}

// conv_w (d,c,3,3) -> (tap, d, c) f16
__global__ void k_pack_conv(const float* __restrict__ w, _Float16* __restrict__ o) {
  int i = blockIdx.x * blockDim.x + threadIdx.x;
  if (i >= 9 * C_ * C_) return;
  int c = i & (C_ - 1);
  int d = (i >> 7) & (C_ - 1);
  int tap = i >> 14;
  o[i] = (_Float16)w[((size_t)(d * C_ + c)) * 9 + tap];
}

// ---------------- LayerNorm over C, CHW layout in; optional f32 CHW + f16 (P,C) out ----
__global__ void k_ln_chw(const float* __restrict__ X, const float* __restrict__ g,
                         const float* __restrict__ b, float* __restrict__ Yc,
                         _Float16* __restrict__ Yp) {
  __shared__ float sm[C_], sq[C_];
  const int p  = blockIdx.x;              // tb*HW + s
  const int tb = p / HW_, s = p - tb * HW_;
  const int c  = threadIdx.x;
  const size_t ix = ((size_t)tb * C_ + c) * HW_ + s;
  float x = X[ix];
  sm[c] = x; sq[c] = x * x;
  __syncthreads();
  for (int st = C_ / 2; st > 0; st >>= 1) {
    if (c < st) { sm[c] += sm[c + st]; sq[c] += sq[c + st]; }
    __syncthreads();
  }
  float mu  = sm[0] * (1.0f / C_);
  float var = sq[0] * (1.0f / C_) - mu * mu;
  float y = (x - mu) * rsqrtf(var + 1e-5f) * g[c] + b[c];
  if (Yc) Yc[ix] = y;
  if (Yp) Yp[(size_t)p * C_ + c] = (_Float16)y;
}

// ---- LN over C on (K,G,HW,C) scan outputs summed over K; emits f16 (P,C) ----
__global__ void k_ln_pc_sum4(const float* __restrict__ Ys, const float* __restrict__ g,
                             const float* __restrict__ b, _Float16* __restrict__ Yp) {
  __shared__ float sm[C_], sq[C_];
  const int p = blockIdx.x;               // g*HW + s
  const int c = threadIdx.x;
  const size_t stride = (size_t)TB_ * HW_ * C_;
  const size_t base   = (size_t)p * C_ + c;
  float x = Ys[base] + Ys[base + stride] + Ys[base + 2 * stride] + Ys[base + 3 * stride];
  sm[c] = x; sq[c] = x * x;
  __syncthreads();
  for (int st = C_ / 2; st > 0; st >>= 1) {
    if (c < st) { sm[c] += sm[c + st]; sq[c] += sq[c + st]; }
    __syncthreads();
  }
  float mu  = sm[0] * (1.0f / C_);
  float var = sq[0] * (1.0f / C_) - mu * mu;
  float y = (x - mu) * rsqrtf(var + 1e-5f) * g[c] + b[c];
  Yp[base] = (_Float16)y;
}

// ---------------- WMMA GEMM: (P,K) f16 row-major  x  (N,K) f16 row-major^T ----------
// One wave computes a 16x32 D tile (two 16x16 accumulators share one A fragment ->
// two independent back-to-back WMMAs per k-step). 4 waves/block cover 128 of N.
// grid = (M/16, N/128).
// A frag: lane row = lane&15, K = k0 + 16*(lane>>4) + {0..15} contiguous.
// B frag: lane col = lane&15, K = k0 + 16*(lane>>4) + {0..15} contiguous.
// D: lane holds rows m0 + r + 8*(lane>>4), col n0 + (lane&15).
template<int EPI>
__global__ void k_gemm(const _Float16* __restrict__ A, const _Float16* __restrict__ Bw,
                       const float* __restrict__ bias,
                       float* __restrict__ outF, _Float16* __restrict__ outH,
                       const float* __restrict__ addsrc,
                       int M, int N, int Kd, int hw) {
  const int wave = threadIdx.x >> 5;
  const int lane = threadIdx.x & 31;
  const int row  = lane & 15;
  const int hi   = lane >> 4;
  const int m0 = blockIdx.x * 16;
  const int n0 = blockIdx.y * 128 + wave * 32;
  const _Float16* Ap  = A  + (size_t)(m0 + row) * Kd + hi * 16;
  const _Float16* Bp0 = Bw + (size_t)(n0 + row) * Kd + hi * 16;
  const _Float16* Bp1 = Bp0 + (size_t)16 * Kd;
  float8 acc0 = {}, acc1 = {};
  for (int k0 = 0; k0 < Kd; k0 += 32) {
    __builtin_prefetch(Ap + k0 + 32, 0, 0);
    half16 a  = mk_a(Ap + k0);
    half16 b0 = *(const half16*)(Bp0 + k0);
    half16 b1 = *(const half16*)(Bp1 + k0);
    acc0 = wmma16(a, b0, acc0);
    acc1 = wmma16(a, b1, acc1);
  }
#pragma unroll
  for (int half_n = 0; half_n < 2; ++half_n) {
    const int n = n0 + half_n * 16 + row;
    const float bn = bias ? bias[n] : 0.0f;
    const float8& acc = half_n ? acc1 : acc0;
#pragma unroll
    for (int r = 0; r < 8; ++r) {
      int m = m0 + r + hi * 8;
      float v = acc[r] + bn;
      if constexpr (EPI == EPI_F16_PC_SILU) v = silu_f(v);
      if constexpr (EPI == EPI_F32_PC)      outF[(size_t)m * N + n] = v;
      if constexpr (EPI == EPI_F16_PC || EPI == EPI_F16_PC_SILU)
        outH[(size_t)m * N + n] = (_Float16)v;
      if constexpr (EPI == EPI_F32_CHW_ADD) {
        int tb = m / hw, s = m - tb * hw;
        size_t o = ((size_t)tb * N + n) * hw + s;
        outF[o] = v + addsrc[o];
      }
    }
  }
}

// ------- LIF over T (tau=2, v_th=1, hard reset): f32 (P,C) in, f16 spikes out -------
// Writes spikes into the interior of a zero-padded (TB, PH, PW, C) buffer so the 3x3
// conv can load every tap unconditionally (EXEC stays all-ones around WMMA).
__global__ void k_lif_spk_pad(const float* __restrict__ X, _Float16* __restrict__ Spad) {
  int idx = blockIdx.x * blockDim.x + threadIdx.x;
  if (idx >= B_ * HW_ * C_) return;
  int bb = idx / (HW_ * C_);
  int j  = idx - bb * (HW_ * C_);
  int c = j & (C_ - 1);
  int s = j >> 7;
  int h = s / W_, w = s - h * W_;
  float v = 0.0f;
#pragma unroll
  for (int t = 0; t < T_; ++t) {
    size_t o = (size_t)(t * B_ + bb) * (HW_ * C_) + j;
    float x = X[o];
    v += (x - v) * 0.5f;
    float sp = (v >= 1.0f) ? 1.0f : 0.0f;
    v *= (1.0f - sp);
    Spad[((size_t)((t * B_ + bb) * PH_ + h + 1) * PW_ + (w + 1)) * C_ + c] = (_Float16)sp;
  }
}

// LIF on y (f32 (P,C)) fused with x1 = xn*(1+spike), output f32 CHW
__global__ void k_lif_combine(const float* __restrict__ Y, const float* __restrict__ Xn,
                              float* __restrict__ X1) {
  int idx = blockIdx.x * blockDim.x + threadIdx.x;
  if (idx >= B_ * HW_ * C_) return;
  int bb = idx / (HW_ * C_);
  int j  = idx - bb * (HW_ * C_);
  int c = j & (C_ - 1);
  int s = j >> 7;
  float v = 0.0f;
#pragma unroll
  for (int t = 0; t < T_; ++t) {
    size_t opc = (size_t)(t * B_ + bb) * (HW_ * C_) + j;
    float x = Y[opc];
    v += (x - v) * 0.5f;
    float sp = (v >= 1.0f) ? 1.0f : 0.0f;
    v *= (1.0f - sp);
    size_t oc = ((size_t)((t * B_ + bb) * C_ + c)) * HW_ + s;
    X1[oc] = Xn[oc] * (1.0f + sp);
  }
}

// ---------------- 3x3 conv 128->128 as implicit WMMA GEMM over 9 taps --------------
// Input is the zero-padded spike buffer: all 9 tap loads are unconditional.
// M tile = 16 consecutive pixels along W (W=48=3*16, no row wrap). One wave computes a
// 16x32 output tile (2 accumulators); 4 waves cover N=128. 72 WMMAs/wave, no branches.
__global__ void k_conv3x3(const _Float16* __restrict__ Xp, const _Float16* __restrict__ Wc,
                          const float* __restrict__ bias, _Float16* __restrict__ Y) {
  const int wave = threadIdx.x >> 5, lane = threadIdx.x & 31;
  const int row = lane & 15, hi = lane >> 4;
  const int tile = blockIdx.x;            // tb*H*3 + h*3 + wt
  const int wt = tile % 3;
  const int h  = (tile / 3) % H_;
  const int tb = tile / (3 * H_);
  const int w0 = wt * 16;
  const int n0 = wave * 32;
  float8 acc0 = {}, acc1 = {};
  // padded coords: output pixel (h, w0+row) tap (dh,dw) -> padded (h+1+dh, w0+row+1+dw)
  const _Float16* Abase =
      Xp + (((size_t)(tb * PH_ + h + 1)) * PW_ + (w0 + row + 1)) * C_ + hi * 16;
  for (int tap = 0; tap < 9; ++tap) {
    const int dh = tap / 3 - 1, dw = tap % 3 - 1;
    const _Float16* Ap  = Abase + ((long)dh * PW_ + dw) * C_;
    const _Float16* Bp0 = Wc + ((size_t)(tap * C_ + n0 + row)) * C_ + hi * 16;
    const _Float16* Bp1 = Bp0 + (size_t)16 * C_;
#pragma unroll
    for (int k0 = 0; k0 < C_; k0 += 32) {
      half16 a  = mk_a(Ap + k0);
      half16 b0 = *(const half16*)(Bp0 + k0);
      half16 b1 = *(const half16*)(Bp1 + k0);
      acc0 = wmma16(a, b0, acc0);           // halo rows are zero == 'SAME' padding
      acc1 = wmma16(a, b1, acc1);
    }
  }
  const size_t pbase = (size_t)(tb * H_ + h) * W_ + w0;
#pragma unroll
  for (int half_n = 0; half_n < 2; ++half_n) {
    const int n = n0 + half_n * 16 + row;
    const float bn = bias[n];
    const float8& acc = half_n ? acc1 : acc0;
#pragma unroll
    for (int r = 0; r < 8; ++r) {
      float v = silu_f(acc[r] + bn);
      Y[(pbase + r + hi * 8) * C_ + n] = (_Float16)v;
    }
  }
}

// ---------------- depthwise 3x3 + silu: f16 (P,C) in, f32 (G,HW,C) out --------------
__global__ void k_dwconv(const _Float16* __restrict__ X, const float* __restrict__ Wd,
                         float* __restrict__ Y) {
  int idx = blockIdx.x * blockDim.x + threadIdx.x;
  if (idx >= TB_ * HW_ * C_) return;
  int c = idx & (C_ - 1);
  int p = idx >> 7;
  int s = p % HW_, gg = p / HW_;
  int h = s / W_, w = s - h * W_;
  float acc = 0.0f;
#pragma unroll
  for (int dh = -1; dh <= 1; ++dh) {
    int hh = h + dh; if (hh < 0 || hh >= H_) continue;
#pragma unroll
    for (int dw = -1; dw <= 1; ++dw) {
      int ww = w + dw; if (ww < 0 || ww >= W_) continue;
      acc += (float)X[((size_t)(gg * HW_ + hh * W_ + ww)) * C_ + c]
             * Wd[c * 9 + (dh + 1) * 3 + (dw + 1)];
    }
  }
  Y[idx] = silu_f(acc);
}

// ---------------- x_proj: 10-vector (dt_r[8], B, C) per (g,k,l) via LDS reduction ---
__global__ void k_xproj(const float* __restrict__ Xss, const float* __restrict__ Wp,
                        float* __restrict__ dbc) {
  __shared__ float red[10][C_];
  const int gkl = blockIdx.x;             // (g*4+k)*HW + l
  const int l  = gkl % HW_;
  const int gk = gkl / HW_;
  const int k = gk & 3, g = gk >> 2;
  const int c = threadIdx.x;
  const int s = map_s(k, l);
  float xv = Xss[((size_t)g * HW_ + s) * C_ + c];
#pragma unroll
  for (int r = 0; r < 10; ++r) red[r][c] = xv * Wp[(k * 10 + r) * C_ + c];
  __syncthreads();
  for (int st = C_ / 2; st > 0; st >>= 1) {
    if (c < st) {
#pragma unroll
      for (int r = 0; r < 10; ++r) red[r][c] += red[r][c + st];
    }
    __syncthreads();
  }
  if (c < 10) dbc[(size_t)gkl * 10 + c] = red[c][0];
}

// ---------------- selective scan: one thread per (g,k,c) chain, L=2304 steps --------
__global__ void k_scan(const float* __restrict__ Xss, const float* __restrict__ dbc,
                       const float* __restrict__ dtw, const float* __restrict__ dtb,
                       const float* __restrict__ Alog, const float* __restrict__ Ds,
                       float* __restrict__ Ys) {
  const int idx = blockIdx.x * blockDim.x + threadIdx.x;   // (g*4+k)*C + c
  if (idx >= TB_ * K_ * C_) return;
  const int c  = idx & (C_ - 1);
  const int gk = idx >> 7;
  const int k = gk & 3, g = gk >> 2;
  float wr[R_];
#pragma unroll
  for (int r = 0; r < R_; ++r) wr[r] = dtw[((size_t)(k * C_ + c)) * R_ + r];
  const float db = dtb[k * C_ + c];
  const float Aa = -__expf(Alog[k * C_ + c]);   // N=1: scalar A per (k,c)
  const float Dv = Ds[k * C_ + c];
  const float* dv = dbc + (size_t)gk * HW_ * 10;
  float hst = 0.0f;
  for (int l = 0; l < HW_; ++l) {
    const float* v = dv + (size_t)l * 10;
    float dtr = db;
#pragma unroll
    for (int r = 0; r < R_; ++r) dtr += v[r] * wr[r];
    float dt = softplus_f(dtr);
    int s = map_s(k, l);
    float xv = Xss[((size_t)g * HW_ + s) * C_ + c];
    hst = __expf(dt * Aa) * hst + dt * v[8] * xv;
    Ys[((size_t)(k * TB_ + g) * HW_ + s) * C_ + c] = hst * v[9] + Dv * xv;  // race-free
  }
}

// =====================================================================================
extern "C" void kernel_launch(void* const* d_in, const int* in_sizes, int n_in,
                              void* d_out, int out_size, void* d_ws, size_t ws_size,
                              hipStream_t stream) {
  (void)in_sizes; (void)n_in; (void)out_size; (void)ws_size;
  const float* x         = (const float*)d_in[0];
  const float* norm1_w   = (const float*)d_in[1];
  const float* norm1_b   = (const float*)d_in[2];
  const float* in_proj_w = (const float*)d_in[3];
  const float* in_proj_b = (const float*)d_in[4];
  const float* conv_w    = (const float*)d_in[5];
  const float* conv_b    = (const float*)d_in[6];
  const float* ss_in_w   = (const float*)d_in[7];
  const float* ss_conv_w = (const float*)d_in[8];
  const float* x_proj_w  = (const float*)d_in[9];
  const float* dt_w      = (const float*)d_in[10];
  const float* dt_b      = (const float*)d_in[11];
  const float* A_log     = (const float*)d_in[12];
  const float* Ds        = (const float*)d_in[13];
  const float* out_nw    = (const float*)d_in[14];
  const float* out_nb    = (const float*)d_in[15];
  const float* ss_out_w  = (const float*)d_in[16];
  const float* norm2_w   = (const float*)d_in[17];
  const float* norm2_b   = (const float*)d_in[18];
  const float* ffn_w1    = (const float*)d_in[19];
  const float* ffn_b1    = (const float*)d_in[20];
  const float* ffn_w2    = (const float*)d_in[21];
  const float* ffn_b2    = (const float*)d_in[22];
  float* out = (float*)d_out;

  char* ws = (char*)d_ws;
  size_t off = 0;
  auto alloc = [&](size_t bytes) -> char* {
    char* p = ws + off; off += (bytes + 255) & ~(size_t)255; return p;
  };
  const size_t PADN = (size_t)TB_ * PH_ * PW_ * C_;              // 5,120,000 halfs
  _Float16* w16_inproj = (_Float16*)alloc((size_t)C_ * C_ * 2);
  _Float16* w16_conv   = (_Float16*)alloc((size_t)9 * C_ * C_ * 2);
  _Float16* w16_ssin   = (_Float16*)alloc((size_t)C_ * C_ * 2);
  _Float16* w16_ssout  = (_Float16*)alloc((size_t)C_ * C_ * 2);
  _Float16* w16_ffn1   = (_Float16*)alloc((size_t)EC_ * C_ * 2);
  _Float16* w16_ffn2   = (_Float16*)alloc((size_t)C_ * EC_ * 2);
  float*    xn_chw     = (float*)   alloc((size_t)P_ * C_ * 4);
  _Float16* act16_a    = (_Float16*)alloc((size_t)P_ * C_ * 2);  // xn16, later x2_16
  float*    pcf32_a    = (float*)   alloc((size_t)P_ * C_ * 4);  // proj out, later y
  _Float16* spk_pad    = (_Float16*)alloc(PADN * 2);             // padded spikes / ln2_16
  _Float16* ln2_16     = spk_pad;                                 // reuse after conv
  char*     convdbc    =            alloc((size_t)P_ * C_ * 2);  // conv16 f16 / dbc f32
  _Float16* conv16     = (_Float16*)convdbc;
  float*    dbc        = (float*)convdbc;                        // 5.9MB <= 9.4MB, serialized
  _Float16* ssin16     = (_Float16*)alloc((size_t)P_ * C_ * 2);
  float*    xss        = (float*)   alloc((size_t)P_ * C_ * 4);
  float*    ys4        = (float*)   alloc((size_t)K_ * P_ * C_ * 4);
  float*    x1_chw     = (float*)   alloc((size_t)P_ * C_ * 4);
  _Float16* ffn1a16    = (_Float16*)alloc((size_t)P_ * EC_ * 2);

  const dim3 b256(256), b128(128);

  // weight packing + halo zeroing (fully rewritten every call; no state carried)
  k_pack_f16<<<dim3((C_ * C_ + 255) / 256),  b256, 0, stream>>>(in_proj_w, w16_inproj, C_ * C_);
  k_pack_conv<<<dim3((9 * C_ * C_ + 255) / 256), b256, 0, stream>>>(conv_w, w16_conv);
  k_pack_f16<<<dim3((C_ * C_ + 255) / 256),  b256, 0, stream>>>(ss_in_w,  w16_ssin,  C_ * C_);
  k_pack_f16<<<dim3((C_ * C_ + 255) / 256),  b256, 0, stream>>>(ss_out_w, w16_ssout, C_ * C_);
  k_pack_f16<<<dim3((EC_ * C_ + 255) / 256), b256, 0, stream>>>(ffn_w1,   w16_ffn1,  EC_ * C_);
  k_pack_f16<<<dim3((C_ * EC_ + 255) / 256), b256, 0, stream>>>(ffn_w2,   w16_ffn2,  C_ * EC_);
  k_zero_h8<<<dim3((int)(PADN / 8 + 255) / 256), b256, 0, stream>>>(spk_pad, (int)(PADN / 8));

  // norm1 -> xn (f32 CHW) + xn16 (f16 P,C)
  k_ln_chw<<<dim3(P_), b128, 0, stream>>>(x, norm1_w, norm1_b, xn_chw, act16_a);
  // in_proj (WMMA)
  k_gemm<EPI_F32_PC><<<dim3(P_ / 16, 1), b128, 0, stream>>>(
      act16_a, w16_inproj, in_proj_b, pcf32_a, nullptr, nullptr, P_, C_, C_, HW_);
  // LIF #1 -> spikes f16 into padded interior
  k_lif_spk_pad<<<dim3((B_ * HW_ * C_ + 255) / 256), b256, 0, stream>>>(pcf32_a, spk_pad);
  // 3x3 conv + bias + silu (implicit WMMA GEMM, no halo branches)
  k_conv3x3<<<dim3(TB_ * H_ * 3), b128, 0, stream>>>(spk_pad, w16_conv, conv_b, conv16);
  // ss_in projection (WMMA)
  k_gemm<EPI_F16_PC><<<dim3(P_ / 16, 1), b128, 0, stream>>>(
      conv16, w16_ssin, nullptr, nullptr, ssin16, nullptr, P_, C_, C_, HW_);
  // depthwise 3x3 + silu
  k_dwconv<<<dim3((P_ * C_ + 255) / 256), b256, 0, stream>>>(ssin16, ss_conv_w, xss);
  // x_proj 10-vector per (g,k,l)
  k_xproj<<<dim3(TB_ * K_ * HW_), b128, 0, stream>>>(xss, x_proj_w, dbc);
  // selective scan, 4 directions, deterministic scatter into ys4
  k_scan<<<dim3((TB_ * K_ * C_ + 127) / 128), b128, 0, stream>>>(
      xss, dbc, dt_w, dt_b, A_log, Ds, ys4);
  // out_norm on sum over directions -> f16 (P,C)   (spk_pad region reused)
  k_ln_pc_sum4<<<dim3(P_), b128, 0, stream>>>(ys4, out_nw, out_nb, ln2_16);
  // ss_out projection (WMMA) -> y f32 (P,C)
  k_gemm<EPI_F32_PC><<<dim3(P_ / 16, 1), b128, 0, stream>>>(
      ln2_16, w16_ssout, nullptr, pcf32_a, nullptr, nullptr, P_, C_, C_, HW_);
  // LIF #2 fused with x1 = xn*(1+spike) -> f32 CHW
  k_lif_combine<<<dim3((B_ * HW_ * C_ + 255) / 256), b256, 0, stream>>>(pcf32_a, xn_chw, x1_chw);
  // norm2 -> x2 f16 (P,C)
  k_ln_chw<<<dim3(P_), b128, 0, stream>>>(x1_chw, norm2_w, norm2_b, nullptr, act16_a);
  // ffn1: 128->512 + bias + silu (WMMA)
  k_gemm<EPI_F16_PC_SILU><<<dim3(P_ / 16, EC_ / 128), b128, 0, stream>>>(
      act16_a, w16_ffn1, ffn_b1, nullptr, ffn1a16, nullptr, P_, EC_, C_, HW_);
  // ffn2: 512->128 + bias + residual x1 -> d_out (T,B,C,H,W) f32
  k_gemm<EPI_F32_CHW_ADD><<<dim3(P_ / 16, 1), b128, 0, stream>>>(
      ffn1a16, w16_ffn2, ffn_b2, out, nullptr, x1_chw, P_, C_, EC_, HW_);
}
